// VRNN_51531017617716
// MI455X (gfx1250) — matrix operand order repaired and checked by reading, not compiled, not run on hardware
//
#include <hip/hip_runtime.h>

// ---------------------------------------------------------------------------
// VRNN forward for MI455X (gfx1250): bf16 WMMA GEMMs (16x16x32), fp32 accum.
// Weights converted to bf16 in d_ws (52.8MB -> resident in 192MB L2).
// W tiles staged into LDS via CDNA5 async copy (global_load_async_to_lds_b128,
// ASYNCcnt-tracked, double-buffered) -> 8x reduction in per-block W traffic.
// Inner loop: batch 4 LDS fragment loads, then 4 back-to-back WMMAs; last
// K-chunk peeled so the steady-state loop has no conditional waits.
// phi_x = mlp2(x) hoisted out of the T-loop (h-independent).
// ---------------------------------------------------------------------------

typedef __attribute__((ext_vector_type(16))) __bf16 v16bf;
typedef __attribute__((ext_vector_type(8)))  __bf16 v8bf;
typedef __attribute__((ext_vector_type(8)))  float  v8f;

#define DEVINL __device__ __forceinline__

// A-fragment (16x32 MxK bf16): lane l -> row l&15, K = {(l>>4)*8 ..+7} u {+16..+23}
DEVINL v16bf load_fragA(const __bf16* row, int k, int ksA) {
  v8bf lo = *(const v8bf*)(row + k + ksA);
  v8bf hi = *(const v8bf*)(row + k + ksA + 16);
  return __builtin_shufflevector(lo, hi, 0,1,2,3,4,5,6,7,8,9,10,11,12,13,14,15);
}
// B-fragment from LDS (32x16 KxN): lane l -> N col l&15 (= W row), K = {(l>>4)*8 ..+7} u {+16..+23}
DEVINL v16bf load_fragB_lds(const __bf16* rowp) {
  v8bf lo = *(const v8bf*)(rowp);
  v8bf hi = *(const v8bf*)(rowp + 16);
  return __builtin_shufflevector(lo, hi, 0,1,2,3,4,5,6,7,8,9,10,11,12,13,14,15);
}

// LDS chunk layout: 64 W-rows x 32 bf16 (64B), row stride padded to 80B
// (conflict-free ds_load_b128), double buffered: 2 x 5120B.
#define WROW_ELEM 40            // 80B / 2
#define WBUF_ELEM (64 * WROW_ELEM)
#define WBUF_BYTES (WBUF_ELEM * 2)

// C = act( [A0|A1] @ [W0;W1]^T + [b0;b1] )
//   act: 0=linear 1=relu 2=sigmoid 3=(linear for col<nsplit, softplus for col>=nsplit)
//   grid = (N/64, M/128), block = 256 (8 waves; wave w owns rows 16w..16w+15, 4 N-tiles)
__global__ __launch_bounds__(256) void gemm_k(
    const __bf16* __restrict__ A0, int K0,
    const __bf16* __restrict__ A1, int K1,
    const __bf16* __restrict__ W0, const __bf16* __restrict__ W1, int nsplit,
    const float* __restrict__ b0, const float* __restrict__ b1,
    float* __restrict__ Cf, __bf16* __restrict__ Cb,
    int N, int act)
{
  __shared__ __align__(16) __bf16 smem[2 * WBUF_ELEM];

  const int tid  = threadIdx.x;
  const int lane = tid & 31;
  const int wv   = tid >> 5;
  const int r    = lane & 15;
  const int hi   = lane >> 4;
  const int ksA  = hi * 8;
  const int m0   = blockIdx.y * 128 + wv * 16;
  const int ldw  = K0 + K1;
  const int ntb  = blockIdx.x * 4;

  // Per-thread W staging source: thread owns 16B slice (tid&3) of row ntb*16 + tid/4.
  const int nrow = ntb * 16 + (tid >> 2);
  const __bf16* wstage =
      ((nrow < nsplit) ? (W0 + (size_t)nrow * ldw)
                       : (W1 + (size_t)(nrow - nsplit) * ldw)) + (tid & 3) * 8;
  // Per-thread LDS destination (byte address relative to LDS base).
  const unsigned lds_mine =
      (unsigned)(size_t)(&smem[0]) + (unsigned)((tid >> 2) * 80 + (tid & 3) * 16);

  const __bf16* a0row = A0 + (size_t)(m0 + r) * K0;
  const __bf16* a1row = (K1 > 0) ? (A1 + (size_t)(m0 + r) * K1) : A0;

  v8f acc[4] = {};

  const int nk = (K0 + K1) / 32;   // always >= 2 here

  // Compute one staged K-chunk: batch the 4 LDS fragment loads into distinct
  // registers, then issue the 4 WMMAs back-to-back (no per-WMMA dscnt stall).
  auto compute = [&](int kc) {
    const int kglob = kc * 32;
    v16bf af;                      // uniform branch -> EXEC stays all-ones
    if (kglob < K0) af = load_fragA(a0row, kglob, ksA);
    else            af = load_fragA(a1row, kglob - K0, ksA);
    const __bf16* lw = smem + (kc & 1) * WBUF_ELEM;
    v16bf wf[4];
#pragma unroll
    for (int j = 0; j < 4; ++j)
      wf[j] = load_fragB_lds(lw + (j * 16 + r) * WROW_ELEM + hi * 8);
#pragma unroll
    for (int j = 0; j < 4; ++j)
      acc[j] = __builtin_amdgcn_wmma_f32_16x16x32_bf16(false, af, false, wf[j],
                                                       (short)0, acc[j], false, false);
  };

  // Prologue: async-stage chunk 0 into buffer 0.
  {
    unsigned long long ga = (unsigned long long)(size_t)wstage;
    asm volatile("global_load_async_to_lds_b128 %0, %1, off"
                 :: "v"(lds_mine), "v"(ga) : "memory");
  }

  // Steady state: unconditional issue of chunk kc+1, wait for chunk kc.
  for (int kc = 0; kc < nk - 1; ++kc) {
    unsigned long long ga = (unsigned long long)(size_t)(wstage + (kc + 1) * 32);
    unsigned dst = lds_mine + (unsigned)(((kc + 1) & 1) * WBUF_BYTES);
    asm volatile("global_load_async_to_lds_b128 %0, %1, off"
                 :: "v"(dst), "v"(ga) : "memory");
    asm volatile("s_wait_asynccnt 0x1" ::: "memory");  // chunk kc landed (in-order)
    __syncthreads();   // cross-wave visibility of chunk kc
    compute(kc);
    __syncthreads();   // all reads done before buffer kc&1 is overwritten
  }
  // Peeled final chunk: wait for everything, no trailing barrier.
  asm volatile("s_wait_asynccnt 0x0" ::: "memory");
  __syncthreads();
  compute(nk - 1);

  // C/D layout: VGPR i -> row m0 + i + (lane>=16 ? 8 : 0), col = ntile*16 + (lane&15)
  const int rowadd = hi * 8;
#pragma unroll
  for (int j = 0; j < 4; ++j) {
    const int col = (ntb + j) * 16 + r;
    float bias = 0.f;
    if (b0) bias = (col < nsplit) ? b0[col] : b1[col - nsplit];
#pragma unroll
    for (int i = 0; i < 8; ++i) {
      float v = acc[j][i] + bias;
      if (act == 1)      v = fmaxf(v, 0.f);
      else if (act == 2) v = 1.f / (1.f + expf(-v));
      else if (act == 3 && col >= nsplit) v = (v > 20.f) ? v : log1pf(expf(v));
      const size_t idx = (size_t)(m0 + rowadd + i) * (size_t)N + (size_t)col;
      if (Cf) Cf[idx] = v;
      if (Cb) Cb[idx] = (__bf16)v;
    }
  }
}

// ---- small elementwise / reduction kernels --------------------------------

__global__ void cvt_k(__bf16* __restrict__ dst, const float* __restrict__ src, int n) {
  for (int i = blockIdx.x * blockDim.x + threadIdx.x; i < n; i += gridDim.x * blockDim.x)
    dst[i] = (__bf16)src[i];
}

__global__ void init_k(float* out, __bf16* h0, int nh) {
  int i = blockIdx.x * blockDim.x + threadIdx.x;
  if (blockIdx.x == 0 && threadIdx.x < 2) out[threadIdx.x] = 0.f;
  for (int j = i; j < nh; j += gridDim.x * blockDim.x) h0[j] = (__bf16)0.f;
}

// z = eps * std + mean ; stat layout: (B, 2Z) with mean cols [0,Z), std cols [Z,2Z)
__global__ void zsample_k(__bf16* __restrict__ z, const float* __restrict__ eps,
                          const float* __restrict__ stat, int Zd, int n) {
  for (int i = blockIdx.x * blockDim.x + threadIdx.x; i < n; i += gridDim.x * blockDim.x) {
    int b = i / Zd, j = i - b * Zd;
    float m = stat[(size_t)b * 2 * Zd + j];
    float s = stat[(size_t)b * 2 * Zd + Zd + j];
    z[i] = (__bf16)(eps[i] * s + m);
  }
}

// GRU combine: gi/gh are (B,3072) fp32 with [r|z|n] blocks of 1024
__global__ void gru_k(const float* __restrict__ gi, const float* __restrict__ gh,
                      const __bf16* __restrict__ h, __bf16* __restrict__ hn, int n) {
  for (int i = blockIdx.x * blockDim.x + threadIdx.x; i < n; i += gridDim.x * blockDim.x) {
    int b = i >> 10, j = i & 1023;
    size_t base = (size_t)b * 3072;
    float rg = 1.f / (1.f + expf(-(gi[base + j]        + gh[base + j])));
    float zg = 1.f / (1.f + expf(-(gi[base + 1024 + j] + gh[base + 1024 + j])));
    float nn = tanhf(gi[base + 2048 + j] + rg * gh[base + 2048 + j]);
    hn[i] = (__bf16)((1.f - zg) * nn + zg * (float)h[i]);
  }
}

// Both KLD terms for one step; single block -> deterministic; out[0] += 0.5*sum/T
__global__ void kld_k(const float* __restrict__ encstat, const float* __restrict__ priorstat,
                      const float* __restrict__ e1stat,  const float* __restrict__ p2stat,
                      float* out) {
  __shared__ float red[256];
  const int tid = threadIdx.x;
  float s = 0.f;
  for (int i = tid; i < 128 * 128; i += 256) {   // level 1: Z=128, stat ld=256
    int b = i >> 7, j = i & 127;
    float m1 = encstat[b * 256 + j],   s1 = encstat[b * 256 + 128 + j];
    float m2 = priorstat[b * 256 + j], s2 = priorstat[b * 256 + 128 + j];
    float d = m1 - m2;
    s += 2.f * logf(s2) - 2.f * logf(s1) + (s1 * s1 + d * d) / (s2 * s2) - 1.f;
  }
  for (int i = tid; i < 128 * 64; i += 256) {    // level 2: Z2=64, stat ld=128
    int b = i >> 6, j = i & 63;
    float m1 = e1stat[b * 128 + j], s1 = e1stat[b * 128 + 64 + j];
    float m2 = p2stat[b * 128 + j], s2 = p2stat[b * 128 + 64 + j];
    float d = m1 - m2;
    s += 2.f * logf(s2) - 2.f * logf(s1) + (s1 * s1 + d * d) / (s2 * s2) - 1.f;
  }
  red[tid] = s;
  __syncthreads();
  for (int st = 128; st > 0; st >>= 1) {
    if (tid < st) red[tid] += red[tid + st];
    __syncthreads();
  }
  if (tid == 0) out[0] += 0.5f * red[0] * (1.0f / 128.0f);
}

// Bernoulli NLL for one step + pred row; single block; out[1] += -sum/T
__global__ void nll_k(const float* __restrict__ dm, const float* __restrict__ xt,
                      float* out, float* __restrict__ pred) {
  __shared__ float red[256];
  const int tid = threadIdx.x;
  float s = 0.f;
  for (int i = tid; i < 128 * 128; i += 256) {
    float d = dm[i], xv = xt[i];
    s += xv * logf(d) + (1.f - xv) * logf(1.f - d);
  }
  red[tid] = s;
  __syncthreads();
  for (int st = 128; st > 0; st >>= 1) {
    if (tid < st) red[tid] += red[tid + st];
    __syncthreads();
  }
  if (tid == 0) out[1] += -red[0] * (1.0f / 128.0f);
  __syncthreads();
  if (tid < 128) pred[tid] = dm[127 * 128 + tid];  // dec_mean[-1]
}

// ---------------------------------------------------------------------------

extern "C" void kernel_launch(void* const* d_in, const int* in_sizes, int n_in,
                              void* d_out, int out_size, void* d_ws, size_t ws_size,
                              hipStream_t stream) {
  (void)in_sizes; (void)n_in; (void)out_size; (void)ws_size;
  const int T = 128, B = 128, X = 128, H = 1024;

  const float* x      = (const float*)d_in[0];
  const float* eps_z  = (const float*)d_in[1];
  const float* eps_z2 = (const float*)d_in[2];
#define BIAS(i) ((const float*)d_in[(i)])

  // --- workspace arena ---
  char* base = (char*)d_ws;
  size_t off = 0;
  auto alloc = [&](size_t bytes) -> void* {
    off = (off + 255) & ~(size_t)255;
    void* p = base + off;
    off += bytes;
    return p;
  };

  // weight matrices (d_in index, element count), bf16 copies in ws
  enum { W_PHIX1, W_PHIX2, W_PHIZ, W_PHIZ1, W_ENC1, W_ENC2, W_ENCM, W_ENCS,
         W_E11, W_E12, W_EM1, W_ES1, W_PRIOR, W_PM, W_PS, W_PR2, W_PM2, W_PS2,
         W_D11, W_D12, W_DE1, W_DE2, W_DM, W_GIH, W_GHH, NW };
  const int    widx[NW]  = {3,5,7,9,11,13,15,17,19,21,23,25,27,29,31,33,35,37,39,41,43,45,47,49,50};
  const size_t welem[NW] = {
    1024u*128, 1024u*1024, 1024u*128, 1024u*64,
    1024u*2048, 1024u*1024, 128u*1024, 128u*1024,
    1024u*2048, 1024u*1024, 64u*1024, 64u*1024,
    1024u*1024, 128u*1024, 128u*1024, 1024u*1024,
    64u*1024, 64u*1024,
    1024u*2048, 1024u*1024, 1024u*2048, 1024u*1024, 128u*1024,
    3072u*2048, 3072u*1024 };
  __bf16* WB[NW];
  for (int i = 0; i < NW; ++i) WB[i] = (__bf16*)alloc(welem[i] * 2);

  __bf16* xb    = (__bf16*)alloc((size_t)T * B * X * 2);
  __bf16* phixt = (__bf16*)alloc((size_t)T * B * H * 2);  // mlp2 layer-1 out
  __bf16* phix  = (__bf16*)alloc((size_t)T * B * H * 2);  // phi_x for all t
  __bf16* h0    = (__bf16*)alloc((size_t)B * H * 2);
  __bf16* h1    = (__bf16*)alloc((size_t)B * H * 2);
  __bf16* ench1 = (__bf16*)alloc((size_t)B * H * 2);
  __bf16* ench  = (__bf16*)alloc((size_t)B * H * 2);
  __bf16* priorh= (__bf16*)alloc((size_t)B * H * 2);
  __bf16* phiz  = (__bf16*)alloc((size_t)B * H * 2);
  __bf16* pr2   = (__bf16*)alloc((size_t)B * H * 2);
  __bf16* e1h1  = (__bf16*)alloc((size_t)B * H * 2);
  __bf16* e1h   = (__bf16*)alloc((size_t)B * H * 2);
  __bf16* phiz1 = (__bf16*)alloc((size_t)B * H * 2);
  __bf16* d1h1  = (__bf16*)alloc((size_t)B * H * 2);
  __bf16* d1h   = (__bf16*)alloc((size_t)B * H * 2);
  __bf16* dh1   = (__bf16*)alloc((size_t)B * H * 2);
  __bf16* dh    = (__bf16*)alloc((size_t)B * H * 2);
  __bf16* zb    = (__bf16*)alloc((size_t)B * 128 * 2);
  __bf16* z1b   = (__bf16*)alloc((size_t)B * 64 * 2);
  float* encstat  = (float*)alloc((size_t)B * 256 * 4);
  float* priorstat= (float*)alloc((size_t)B * 256 * 4);
  float* e1stat   = (float*)alloc((size_t)B * 128 * 4);
  float* p2stat   = (float*)alloc((size_t)B * 128 * 4);
  float* decmean  = (float*)alloc((size_t)B * 128 * 4);
  float* gi       = (float*)alloc((size_t)B * 3072 * 4);
  float* gh       = (float*)alloc((size_t)B * 3072 * 4);

  float* out = (float*)d_out;

  auto G = [&](const __bf16* A0, int K0, const __bf16* A1, int K1,
               const __bf16* Wa, const __bf16* Wb2, int nsplit,
               const float* ba, const float* bb,
               float* Cf, __bf16* Cb, int M, int N, int act) {
    gemm_k<<<dim3(N / 64, M / 128), 256, 0, stream>>>(
        A0, K0, A1, K1, Wa, Wb2, nsplit, ba, bb, Cf, Cb, N, act);
  };
  auto CVT = [&](__bf16* dst, const float* src, size_t n) {
    int g = (int)((n + 255) / 256); if (g > 2048) g = 2048;
    cvt_k<<<dim3(g), 256, 0, stream>>>(dst, src, (int)n);
  };

  // ---- init + weight conversion (every call; deterministic) ----
  init_k<<<dim3(512), 256, 0, stream>>>(out, h0, B * H);
  for (int i = 0; i < NW; ++i) CVT(WB[i], (const float*)d_in[widx[i]], welem[i]);
  CVT(xb, x, (size_t)T * B * X);

  // ---- hoisted phi_x = mlp2(x) over all T*B rows ----
  G(xb, 128,  nullptr, 0, WB[W_PHIX1], WB[W_PHIX1], H, BIAS(4), nullptr, nullptr, phixt, T * B, H, 1);
  G(phixt, H, nullptr, 0, WB[W_PHIX2], WB[W_PHIX2], H, BIAS(6), nullptr, nullptr, phix,  T * B, H, 1);

  // ---- sequential scan over T ----
  for (int t = 0; t < T; ++t) {
    const __bf16* hc = (t & 1) ? h1 : h0;
    __bf16*       hn = (t & 1) ? h0 : h1;
    const __bf16* pxt = phix + (size_t)t * B * H;

    // enc branch: enc_h = mlp2([phi_x, h])
    G(pxt, H, hc, H, WB[W_ENC1], WB[W_ENC1], H, BIAS(12), nullptr, nullptr, ench1, B, H, 1);
    G(ench1, H, nullptr, 0, WB[W_ENC2], WB[W_ENC2], H, BIAS(14), nullptr, nullptr, ench, B, H, 1);
    G(ench, H, nullptr, 0, WB[W_ENCM], WB[W_ENCS], 128, BIAS(16), BIAS(18), encstat, nullptr, B, 256, 3);
    // prior branch
    G(hc, H, nullptr, 0, WB[W_PRIOR], WB[W_PRIOR], H, BIAS(28), nullptr, nullptr, priorh, B, H, 1);
    G(priorh, H, nullptr, 0, WB[W_PM], WB[W_PS], 128, BIAS(30), BIAS(32), priorstat, nullptr, B, 256, 3);
    // z sample -> phi_z -> prior2
    zsample_k<<<dim3(64), 256, 0, stream>>>(zb, eps_z + (size_t)t * B * 128, encstat, 128, B * 128);
    G(zb, 128, nullptr, 0, WB[W_PHIZ], WB[W_PHIZ], H, BIAS(8), nullptr, nullptr, phiz, B, H, 1);
    G(phiz, H, nullptr, 0, WB[W_PR2], WB[W_PR2], H, BIAS(34), nullptr, nullptr, pr2, B, H, 1);
    G(pr2, H, nullptr, 0, WB[W_PM2], WB[W_PS2], 64, BIAS(36), BIAS(38), p2stat, nullptr, B, 128, 3);
    // enc1 branch: enc1_h = mlp2([phi_z, h])
    G(phiz, H, hc, H, WB[W_E11], WB[W_E11], H, BIAS(20), nullptr, nullptr, e1h1, B, H, 1);
    G(e1h1, H, nullptr, 0, WB[W_E12], WB[W_E12], H, BIAS(22), nullptr, nullptr, e1h, B, H, 1);
    G(e1h, H, nullptr, 0, WB[W_EM1], WB[W_ES1], 64, BIAS(24), BIAS(26), e1stat, nullptr, B, 128, 3);
    // KLD (both levels) -> out[0]
    kld_k<<<dim3(1), 256, 0, stream>>>(encstat, priorstat, e1stat, p2stat, out);
    // z1 sample -> phi_z1
    zsample_k<<<dim3(32), 256, 0, stream>>>(z1b, eps_z2 + (size_t)t * B * 64, e1stat, 64, B * 64);
    G(z1b, 64, nullptr, 0, WB[W_PHIZ1], WB[W_PHIZ1], H, BIAS(10), nullptr, nullptr, phiz1, B, H, 1);
    // decoder: dec1 = mlp2([phi_z1, h]); dec = mlp2([dec1, h]); dec_mean = sig(lin)
    G(phiz1, H, hc, H, WB[W_D11], WB[W_D11], H, BIAS(40), nullptr, nullptr, d1h1, B, H, 1);
    G(d1h1, H, nullptr, 0, WB[W_D12], WB[W_D12], H, BIAS(42), nullptr, nullptr, d1h, B, H, 1);
    G(d1h, H, hc, H, WB[W_DE1], WB[W_DE1], H, BIAS(44), nullptr, nullptr, dh1, B, H, 1);
    G(dh1, H, nullptr, 0, WB[W_DE2], WB[W_DE2], H, BIAS(46), nullptr, nullptr, dh, B, H, 1);
    G(dh, H, nullptr, 0, WB[W_DM], WB[W_DM], 128, BIAS(48), nullptr, decmean, nullptr, B, 128, 2);
    nll_k<<<dim3(1), 256, 0, stream>>>(decmean, x + (size_t)t * B * X, out, out + 2 + (size_t)t * 128);
    // GRU: gi = [phi_x, phi_z1] @ Wih^T ; gh = h @ Whh^T ; combine
    G(pxt, H, phiz1, H, WB[W_GIH], WB[W_GIH], 3072, nullptr, nullptr, gi, nullptr, B, 3072, 0);
    G(hc, H, nullptr, 0, WB[W_GHH], WB[W_GHH], 3072, nullptr, nullptr, gh, nullptr, B, 3072, 0);
    gru_k<<<dim3(512), 256, 0, stream>>>(gi, gh, hc, hn, B * H);
  }
#undef BIAS
}
